// GraphTransformerLayer_11038065951379
// MI455X (gfx1250) — matrix operand order repaired and checked by reading, not compiled, not run on hardware
//
#include <hip/hip_runtime.h>
#include <hip/hip_bf16.h>
#include <math.h>

typedef unsigned short u16;
typedef __bf16 bf16x16 __attribute__((ext_vector_type(16)));
typedef float f32x8 __attribute__((ext_vector_type(8)));

#define B_    2
#define N_    512
#define D_    256
#define DE_   64
#define H_    8
#define DH_   32
#define DFF_  1024
#define DEFF_ 256
#define NN_   (N_*N_)      /* 262144 */
#define NE_   (B_*NN_)     /* 524288 */
#define MX_   (B_*N_)      /* 1024  */
#define EPS_  1e-5f

// ---------------- helpers ----------------
__device__ inline u16 f2bf(float f) {
  unsigned u = __float_as_uint(f);
  u += 0x7FFFu + ((u >> 16) & 1u);          // round-to-nearest-even
  return (u16)(u >> 16);
}
__device__ inline unsigned pack2(float a, float b) {
  return (unsigned)f2bf(a) | ((unsigned)f2bf(b) << 16);
}
__device__ inline float gelu_exact(float v) {
  return 0.5f * v * (1.0f + erff(v * 0.7071067811865475f));
}
__device__ inline f32x8 zero8() {
  f32x8 z = {0.f,0.f,0.f,0.f,0.f,0.f,0.f,0.f};
  return z;
}

union FragU { bf16x16 v; uint4 q[2]; };

// A fragment (16x32 bf16): lane row = r, K[k0+hsel*16 .. +16) contiguous
__device__ inline bf16x16 load_fragA(const u16* p, int ld, int row, int k0, int hsel) {
  FragU f;
  const uint4* s = (const uint4*)(p + (long)row*ld + k0 + hsel*16);
  f.q[0] = s[0];
  f.q[1] = s[1];
  return f.v;
}
// B fragment (32x16 bf16) from BT row-major [N,K]: lane col = c, K[k0+hsel*16 .. +16) contiguous
__device__ inline bf16x16 load_fragB(const u16* p, int ld, int col, int k0, int hsel) {
  FragU f;
  const uint4* s = (const uint4*)(p + (long)col*ld + k0 + hsel*16);
  f.q[0] = s[0];
  f.q[1] = s[1];
  return f.v;
}
__device__ inline f32x8 wmma_bf16(bf16x16 a, bf16x16 b, f32x8 c) {
  return __builtin_amdgcn_wmma_f32_16x16x32_bf16(false, a, false, b, (short)0, c, false, false);
}

// ---------------- weight transpose + bf16 convert:  WT[n,k] = bf16(W[k,n]) ----------------
__global__ __launch_bounds__(256) void w2bfT_k(const float* __restrict__ W,
                                               u16* __restrict__ WT, int K, int Nc) {
  const long i = (long)blockIdx.x*256 + threadIdx.x;
  if (i >= (long)K*Nc) return;
  const int n = (int)(i % Nc);
  const int k = (int)(i / Nc);
  WT[(long)n*K + k] = f2bf(W[i]);
}

// ---------------- LayerNorm over rows of length 256 -> bf16 ----------------
__global__ __launch_bounds__(256) void ln256_bf16_k(const float* __restrict__ in,
                                                    const float* __restrict__ g,
                                                    const float* __restrict__ bb,
                                                    u16* __restrict__ out, int rows) {
  const int lane = threadIdx.x & 31, wave = threadIdx.x >> 5;
  const long row = (long)blockIdx.x*8 + wave;
  if (row >= rows) return;
  const float* p = in + row*D_ + lane*8;
  float4 x0 = ((const float4*)p)[0];
  float4 x1 = ((const float4*)p)[1];
  float s = x0.x+x0.y+x0.z+x0.w + x1.x+x1.y+x1.z+x1.w;
#pragma unroll
  for (int o = 16; o; o >>= 1) s += __shfl_xor(s, o, 32);
  const float mean = s * (1.f/D_);
  float d, vs = 0.f;
  d=x0.x-mean; vs+=d*d; d=x0.y-mean; vs+=d*d; d=x0.z-mean; vs+=d*d; d=x0.w-mean; vs+=d*d;
  d=x1.x-mean; vs+=d*d; d=x1.y-mean; vs+=d*d; d=x1.z-mean; vs+=d*d; d=x1.w-mean; vs+=d*d;
#pragma unroll
  for (int o = 16; o; o >>= 1) vs += __shfl_xor(vs, o, 32);
  const float inv = rsqrtf(vs*(1.f/D_) + EPS_);
  const float4 g0 = ((const float4*)(g  + lane*8))[0];
  const float4 g1 = ((const float4*)(g  + lane*8))[1];
  const float4 b0 = ((const float4*)(bb + lane*8))[0];
  const float4 b1 = ((const float4*)(bb + lane*8))[1];
  uint4 o4;
  o4.x = pack2((x0.x-mean)*inv*g0.x + b0.x, (x0.y-mean)*inv*g0.y + b0.y);
  o4.y = pack2((x0.z-mean)*inv*g0.z + b0.z, (x0.w-mean)*inv*g0.w + b0.w);
  o4.z = pack2((x1.x-mean)*inv*g1.x + b1.x, (x1.y-mean)*inv*g1.y + b1.y);
  o4.w = pack2((x1.z-mean)*inv*g1.z + b1.z, (x1.w-mean)*inv*g1.w + b1.w);
  ((uint4*)(out + row*D_ + lane*8))[0] = o4;
}

// ---------------- edge bias: LN(e[edge,:64]) @ we[64,8] + be -> scores[b,h,i,j] ----------------
__global__ __launch_bounds__(256) void ebias_k(const float* __restrict__ e,
                                               const float* __restrict__ g,
                                               const float* __restrict__ bb,
                                               const float* __restrict__ we,
                                               const float* __restrict__ be,
                                               float* __restrict__ scores) {
  const int lane = threadIdx.x & 31, wave = threadIdx.x >> 5;
  const long edge = (long)blockIdx.x*8 + wave;
  if (edge >= NE_) return;
  const float2 x = ((const float2*)(e + edge*DE_))[lane];
  float s = x.x + x.y;
#pragma unroll
  for (int o = 16; o; o >>= 1) s += __shfl_xor(s, o, 32);
  const float mean = s * (1.f/DE_);
  const float d0 = x.x - mean, d1 = x.y - mean;
  float vs = d0*d0 + d1*d1;
#pragma unroll
  for (int o = 16; o; o >>= 1) vs += __shfl_xor(vs, o, 32);
  const float inv = rsqrtf(vs*(1.f/DE_) + EPS_);
  const int c0 = lane*2;
  const float y0 = d0*inv*g[c0]   + bb[c0];
  const float y1 = d1*inv*g[c0+1] + bb[c0+1];
  float h[H_];
#pragma unroll
  for (int hh = 0; hh < H_; ++hh)
    h[hh] = y0*we[c0*H_ + hh] + y1*we[(c0+1)*H_ + hh];
#pragma unroll
  for (int hh = 0; hh < H_; ++hh)
#pragma unroll
    for (int o = 16; o; o >>= 1) h[hh] += __shfl_xor(h[hh], o, 32);
  const long bidx = edge / NN_;
  const long ij   = edge - bidx*NN_;
  float val = h[0];
#pragma unroll
  for (int hh = 1; hh < H_; ++hh) val = (lane == hh) ? h[hh] : val;
  if (lane < H_)
    scores[(bidx*H_ + lane)*NN_ + ij] = val + be[lane];
}

// ---------------- generic bf16 WMMA GEMM (double-buffered fragments) ----------------
// C = act((A@B + bias) * outScale) + Res
// A: bf16 row-major [M,K] (lda), BT: bf16 row-major [N,K] (ldb)
// OM: 0 = f32 row-major, 1 = bf16 row-major, 2 = bf16 transposed (C[col*ldc+row])
template<int OM, int ACT, bool RES, int CT>
__global__ __launch_bounds__(256) void gemm_bf16_k(
    const u16* __restrict__ A, int lda, long sA1, long sA2,
    const u16* __restrict__ BT, int ldb, long sB1, long sB2,
    float* Cf, u16* Cb, int ldc, long sC1, long sC2,
    const float* __restrict__ bias,
    const float* Res, int ldr, long sR1, long sR2,
    int M, int N, int K, int HB, float outScale) {
  const int z  = blockIdx.z;
  const int zb = z / HB, zh = z % HB;
  const u16* Ab = A + zb*sA1 + zh*sA2;
  const u16* Bb = BT + zb*sB1 + zh*sB2;
  const long cbase = zb*sC1 + zh*sC2;
  const int lane = threadIdx.x & 31;
  const int wave = threadIdx.x >> 5;
  const int row0 = (blockIdx.y*8 + wave)*16;
  const int col0 = blockIdx.x*(16*CT);
  if (row0 >= M) return;
  const int r = lane & 15, hs = lane >> 4;

  f32x8 acc[CT];
#pragma unroll
  for (int t = 0; t < CT; ++t) acc[t] = zero8();

  // prologue loads (k-step 0)
  bf16x16 a = load_fragA(Ab, lda, row0 + r, 0, hs);
  bf16x16 b[CT];
#pragma unroll
  for (int t = 0; t < CT; ++t)
    b[t] = load_fragB(Bb, ldb, col0 + t*16 + r, 0, hs);

  for (int k0 = 0; k0 < K; k0 += 32) {
    // issue next k-step's loads (clamped on last iter; overlaps with WMMAs below)
    const int kn = (k0 + 32 < K) ? (k0 + 32) : k0;
    bf16x16 an = load_fragA(Ab, lda, row0 + r, kn, hs);
    bf16x16 bn[CT];
#pragma unroll
    for (int t = 0; t < CT; ++t)
      bn[t] = load_fragB(Bb, ldb, col0 + t*16 + r, kn, hs);
#pragma unroll
    for (int t = 0; t < CT; ++t)
      acc[t] = wmma_bf16(a, b[t], acc[t]);
    a = an;
#pragma unroll
    for (int t = 0; t < CT; ++t) b[t] = bn[t];
  }

#pragma unroll
  for (int t = 0; t < CT; ++t) {
    const int cc = col0 + t*16 + r;
    const float bval = bias ? bias[cc] : 0.f;
#pragma unroll
    for (int j = 0; j < 8; ++j) {
      const int rr = row0 + hs*8 + j;
      float o = (acc[t][j] + bval) * outScale;
      if (ACT == 1) o = gelu_exact(o);
      if (RES) o += Res[zb*sR1 + zh*sR2 + (long)rr*ldr + cc];
      if (OM == 0)      Cf[cbase + (long)rr*ldc + cc] = o;
      else if (OM == 1) Cb[cbase + (long)rr*ldc + cc] = f2bf(o);
      else              Cb[cbase + (long)cc*ldc + rr] = f2bf(o);
    }
  }
}

// ---------------- softmax over rows of 512, out bf16 ----------------
__global__ __launch_bounds__(256) void softmax_k(const float* __restrict__ sc,
                                                 u16* __restrict__ attn) {
  const int lane = threadIdx.x & 31, wave = threadIdx.x >> 5;
  const long row = (long)blockIdx.x*8 + wave;
  if (row >= (long)B_*H_*N_) return;
  const float* p = sc + row*N_ + lane*16;
  float4 v[4];
#pragma unroll
  for (int i = 0; i < 4; ++i) v[i] = ((const float4*)p)[i];
  float m = -3.4e38f;
#pragma unroll
  for (int i = 0; i < 4; ++i)
    m = fmaxf(m, fmaxf(fmaxf(v[i].x, v[i].y), fmaxf(v[i].z, v[i].w)));
#pragma unroll
  for (int o = 16; o; o >>= 1) m = fmaxf(m, __shfl_xor(m, o, 32));
  float s = 0.f;
#pragma unroll
  for (int i = 0; i < 4; ++i) {
    v[i].x = __expf(v[i].x - m); v[i].y = __expf(v[i].y - m);
    v[i].z = __expf(v[i].z - m); v[i].w = __expf(v[i].w - m);
    s += v[i].x + v[i].y + v[i].z + v[i].w;
  }
#pragma unroll
  for (int o = 16; o; o >>= 1) s += __shfl_xor(s, o, 32);
  const float inv = 1.f / s;
  uint4 o0, o1;
  o0.x = pack2(v[0].x*inv, v[0].y*inv); o0.y = pack2(v[0].z*inv, v[0].w*inv);
  o0.z = pack2(v[1].x*inv, v[1].y*inv); o0.w = pack2(v[1].z*inv, v[1].w*inv);
  o1.x = pack2(v[2].x*inv, v[2].y*inv); o1.y = pack2(v[2].z*inv, v[2].w*inv);
  o1.z = pack2(v[3].x*inv, v[3].y*inv); o1.w = pack2(v[3].z*inv, v[3].w*inv);
  uint4* q = (uint4*)(attn + row*N_ + lane*16);
  q[0] = o0; q[1] = o1;
}

// ---------------- out_e (pre-FFN) = scores^T @ weo + beo + e ----------------
__global__ __launch_bounds__(256) void out_e_k(const float* __restrict__ sc,
                                               const float* __restrict__ e,
                                               const float* __restrict__ weo,
                                               const float* __restrict__ beo,
                                               float* __restrict__ oute) {
  const long idx = (long)blockIdx.x*256 + threadIdx.x;
  if (idx >= (long)NE_*DE_) return;
  const int de = (int)(idx & (DE_-1));
  const long edge = idx >> 6;
  const long bidx = edge / NN_;
  const long ij   = edge - bidx*NN_;
  float s = beo[de];
#pragma unroll
  for (int hh = 0; hh < H_; ++hh)
    s += sc[(bidx*H_ + hh)*NN_ + ij] * weo[hh*DE_ + de];
  oute[idx] = s + e[idx];
}

// ---------------- fused edge FFN: out_e = GELU(LN(out_e)@w1+b1)@w2 + b2 + out_e (in place) ----
// One wave handles 16 edges.  w1T staged in LDS once per block; per-wave LDS slab for
// LN output + C->A repack of the GELU'd hidden chunk.
__global__ __launch_bounds__(256) void edge_ffn_k(const float* __restrict__ g,
                                                  const float* __restrict__ bvec,
                                                  const u16* __restrict__ w1T,  // [256,64] bf16
                                                  const float* __restrict__ b1,
                                                  const u16* __restrict__ w2T,  // [64,256] bf16
                                                  const float* __restrict__ b2,
                                                  float* oute) {
  __shared__ __align__(16) u16 w1s[DEFF_*DE_];  // 32 KB staged w1T
  __shared__ __align__(16) u16 lds[8*1536];     // per wave: 16x64 he + 16x32 repack
  // ---- cooperative stage of w1T into LDS (2048 x uint4) ----
  {
    const uint4* srcw = (const uint4*)w1T;
    uint4* dstw = (uint4*)w1s;
#pragma unroll
    for (int i = 0; i < 8; ++i)
      dstw[threadIdx.x + i*256] = srcw[threadIdx.x + i*256];
  }
  __syncthreads();

  const int lane = threadIdx.x & 31, wave = threadIdx.x >> 5;
  u16* he = lds + wave*1536;
  u16* rp = he + 1024;
  const long rowBase = ((long)blockIdx.x*8 + wave)*16;
  if (rowBase >= NE_) return;
  const int r = lane & 15, hs = lane >> 4;

  // --- LayerNorm of 16 rows x 64: lane handles (row r, half hs: 32 contiguous floats) ---
  const float* src = oute + (rowBase + r)*DE_ + hs*32;
  float4 x[8];
#pragma unroll
  for (int i = 0; i < 8; ++i) x[i] = ((const float4*)src)[i];
  float s = 0.f;
#pragma unroll
  for (int i = 0; i < 8; ++i) s += x[i].x + x[i].y + x[i].z + x[i].w;
  s += __shfl_xor(s, 16, 32);
  const float mean = s * (1.f/DE_);
  float vs = 0.f;
#pragma unroll
  for (int i = 0; i < 8; ++i) {
    float a0 = x[i].x-mean, a1 = x[i].y-mean, a2 = x[i].z-mean, a3 = x[i].w-mean;
    vs += a0*a0 + a1*a1 + a2*a2 + a3*a3;
  }
  vs += __shfl_xor(vs, 16, 32);
  const float inv = rsqrtf(vs*(1.f/DE_) + EPS_);
  const float* gp = g    + hs*32;
  const float* bp = bvec + hs*32;
  unsigned ow[16];
#pragma unroll
  for (int i = 0; i < 8; ++i) {
    float4 gv = ((const float4*)gp)[i];
    float4 bv = ((const float4*)bp)[i];
    ow[2*i]   = pack2((x[i].x-mean)*inv*gv.x + bv.x, (x[i].y-mean)*inv*gv.y + bv.y);
    ow[2*i+1] = pack2((x[i].z-mean)*inv*gv.z + bv.z, (x[i].w-mean)*inv*gv.w + bv.w);
  }
  uint4* dst = (uint4*)(he + r*64 + hs*32);
#pragma unroll
  for (int i = 0; i < 4; ++i)
    dst[i] = make_uint4(ow[4*i], ow[4*i+1], ow[4*i+2], ow[4*i+3]);

  // invariant A fragments of the first GEMM (same-wave LDS ops are in-order)
  const bf16x16 a0 = load_fragA(he, 64, r, 0,  hs);
  const bf16x16 a1 = load_fragA(he, 64, r, 32, hs);

  // --- GEMM1 (K=64, w1 from LDS) -> GELU -> LDS repack -> GEMM2 (K chunks of 32) ---
  f32x8 acc[4];
#pragma unroll
  for (int t = 0; t < 4; ++t) acc[t] = zero8();

#pragma unroll 1
  for (int pp = 0; pp < 8; ++pp) {
    // prefetch this chunk's w2 B-fragments from global; latency hides under GEMM1
    bf16x16 b2v[4];
#pragma unroll
    for (int c2 = 0; c2 < 4; ++c2)
      b2v[c2] = load_fragB(w2T, DEFF_, c2*16 + r, pp*32, hs);

#pragma unroll
    for (int t = 0; t < 2; ++t) {
      const int n = 2*pp + t;
      f32x8 c1 = zero8();
      c1 = wmma_bf16(a0, load_fragB(w1s, 64, n*16 + r, 0,  hs), c1);
      c1 = wmma_bf16(a1, load_fragB(w1s, 64, n*16 + r, 32, hs), c1);
      const float bb1 = b1[n*16 + r];
#pragma unroll
      for (int j = 0; j < 8; ++j) {
        const float v = gelu_exact(c1[j] + bb1);
        rp[(hs*8 + j)*32 + t*16 + r] = f2bf(v);   // C-layout -> row-major bf16 in LDS
      }
    }
    // per-wave private LDS; same-wave DS ops are in-order, no barrier needed
    bf16x16 a2 = load_fragA(rp, 32, r, 0, hs);
#pragma unroll
    for (int c2 = 0; c2 < 4; ++c2)
      acc[c2] = wmma_bf16(a2, b2v[c2], acc[c2]);
  }

  // --- epilogue: + b2 + residual, in place ---
#pragma unroll
  for (int c2 = 0; c2 < 4; ++c2) {
    const int cc = c2*16 + r;
    const float bb2 = b2[cc];
#pragma unroll
    for (int j = 0; j < 8; ++j) {
      const long rr = rowBase + hs*8 + j;
      oute[rr*DE_ + cc] = acc[c2][j] + bb2 + oute[rr*DE_ + cc];
    }
  }
}

// ---------------- host ----------------
extern "C" void kernel_launch(void* const* d_in, const int* in_sizes, int n_in,
                              void* d_out, int out_size, void* d_ws, size_t ws_size,
                              hipStream_t stream) {
  (void)in_sizes; (void)n_in; (void)out_size; (void)ws_size;
  const float* x        = (const float*)d_in[0];
  const float* e        = (const float*)d_in[1];
  const float* ln_x_g   = (const float*)d_in[2];
  const float* ln_x_b   = (const float*)d_in[3];
  const float* ln_e_g   = (const float*)d_in[4];
  const float* ln_e_b   = (const float*)d_in[5];
  const float* wq = (const float*)d_in[6];   const float* bq = (const float*)d_in[7];
  const float* wk = (const float*)d_in[8];   const float* bk = (const float*)d_in[9];
  const float* wv = (const float*)d_in[10];  const float* bv = (const float*)d_in[11];
  const float* wo = (const float*)d_in[12];  const float* bo = (const float*)d_in[13];
  const float* we = (const float*)d_in[14];  const float* be = (const float*)d_in[15];
  const float* weo = (const float*)d_in[16]; const float* beo = (const float*)d_in[17];
  const float* ffx_ln_g = (const float*)d_in[18]; const float* ffx_ln_b = (const float*)d_in[19];
  const float* ffx_w1 = (const float*)d_in[20];   const float* ffx_b1 = (const float*)d_in[21];
  const float* ffx_w2 = (const float*)d_in[22];   const float* ffx_b2 = (const float*)d_in[23];
  const float* ffe_ln_g = (const float*)d_in[24]; const float* ffe_ln_b = (const float*)d_in[25];
  const float* ffe_w1 = (const float*)d_in[26];   const float* ffe_b1 = (const float*)d_in[27];
  const float* ffe_w2 = (const float*)d_in[28];   const float* ffe_b2 = (const float*)d_in[29];

  float* out_x = (float*)d_out;
  float* out_e = out_x + (long)MX_*D_;

  char* wsp = (char*)d_ws;
  size_t off = 0;
  auto alloc = [&](size_t bytes) -> char* {
    off = (off + 255) & ~(size_t)255;
    char* p = wsp + off; off += bytes; return p;
  };
  u16* wqT  = (u16*)alloc((size_t)D_*D_*2);
  u16* wkT  = (u16*)alloc((size_t)D_*D_*2);
  u16* wvT  = (u16*)alloc((size_t)D_*D_*2);
  u16* woT  = (u16*)alloc((size_t)D_*D_*2);
  u16* fw1T = (u16*)alloc((size_t)D_*DFF_*2);
  u16* fw2T = (u16*)alloc((size_t)D_*DFF_*2);
  u16* ew1T = (u16*)alloc((size_t)DE_*DEFF_*2);
  u16* ew2T = (u16*)alloc((size_t)DE_*DEFF_*2);
  u16* xn   = (u16*)alloc((size_t)MX_*D_*2);
  u16* qb   = (u16*)alloc((size_t)MX_*D_*2);
  u16* kb   = (u16*)alloc((size_t)MX_*D_*2);
  u16* vT   = (u16*)alloc((size_t)MX_*D_*2);       // [B,H,DH,N] bf16
  float* scores = (float*)alloc((size_t)B_*H_*NN_*4);
  u16* attn = (u16*)alloc((size_t)B_*H_*NN_*2);
  u16* ctx  = (u16*)alloc((size_t)MX_*D_*2);
  float* oxp = (float*)alloc((size_t)MX_*D_*4);
  u16* hx   = (u16*)alloc((size_t)MX_*D_*2);
  u16* h1   = (u16*)alloc((size_t)MX_*DFF_*2);

  dim3 blk(256);

  // 1) weights -> transposed bf16
  w2bfT_k<<<D_*D_/256, blk, 0, stream>>>(wq, wqT, D_, D_);
  w2bfT_k<<<D_*D_/256, blk, 0, stream>>>(wk, wkT, D_, D_);
  w2bfT_k<<<D_*D_/256, blk, 0, stream>>>(wv, wvT, D_, D_);
  w2bfT_k<<<D_*D_/256, blk, 0, stream>>>(wo, woT, D_, D_);
  w2bfT_k<<<D_*DFF_/256, blk, 0, stream>>>(ffx_w1, fw1T, D_,   DFF_);
  w2bfT_k<<<D_*DFF_/256, blk, 0, stream>>>(ffx_w2, fw2T, DFF_, D_);
  w2bfT_k<<<DE_*DEFF_/256, blk, 0, stream>>>(ffe_w1, ew1T, DE_,   DEFF_);
  w2bfT_k<<<DE_*DEFF_/256, blk, 0, stream>>>(ffe_w2, ew2T, DEFF_, DE_);

  // 2) LN(x) -> bf16
  ln256_bf16_k<<<MX_/8, blk, 0, stream>>>(x, ln_x_g, ln_x_b, xn, MX_);

  // 3) edge bias -> scores buffer [B,H,N,N]
  ebias_k<<<NE_/8, blk, 0, stream>>>(e, ln_e_g, ln_e_b, we, be, scores);

  // 4) q (pre-scaled by 1/sqrt(DH)), k -> bf16 [B*N, D]
  gemm_bf16_k<1,0,false,4><<<dim3(D_/64, MX_/128, 1), blk, 0, stream>>>(
      xn, D_, 0, 0, wqT, D_, 0, 0, nullptr, qb, D_, 0, 0, bq,
      nullptr, 0, 0, 0, MX_, D_, D_, 1, 0.17677669529663687f);
  gemm_bf16_k<1,0,false,4><<<dim3(D_/64, MX_/128, 1), blk, 0, stream>>>(
      xn, D_, 0, 0, wkT, D_, 0, 0, nullptr, kb, D_, 0, 0, bk,
      nullptr, 0, 0, 0, MX_, D_, D_, 1, 1.0f);
  // 5) v, stored transposed per batch: vT[b][h*32+d][j]
  gemm_bf16_k<2,0,false,4><<<dim3(D_/64, N_/128, B_), blk, 0, stream>>>(
      xn, D_, (long)N_*D_, 0, wvT, D_, 0, 0, nullptr, vT, N_, (long)D_*N_, 0, bv,
      nullptr, 0, 0, 0, N_, D_, D_, 1, 1.0f);

  // 6) scores = ebias + q @ k^T   (per b,h; in-place residual)
  gemm_bf16_k<0,0,true,4><<<dim3(N_/64, N_/128, B_*H_), blk, 0, stream>>>(
      qb, D_, (long)N_*D_, DH_, kb, D_, (long)N_*D_, DH_,
      scores, nullptr, N_, (long)H_*NN_, (long)NN_, nullptr,
      scores, N_, (long)H_*NN_, (long)NN_, N_, N_, DH_, H_, 1.0f);

  // 7) out_e (pre-FFN) -> d_out edge region
  out_e_k<<<(int)((long)NE_*DE_/256), blk, 0, stream>>>(scores, e, weo, beo, out_e);

  // 8) softmax -> attn bf16
  softmax_k<<<(B_*H_*N_)/8, blk, 0, stream>>>(scores, attn);

  // 9) ctx = attn @ v  -> bf16 [B,N,H,DH] = [B*N, D]
  gemm_bf16_k<1,0,false,2><<<dim3(1, N_/128, B_*H_), blk, 0, stream>>>(
      attn, N_, (long)H_*NN_, (long)NN_, vT, N_, (long)H_*DH_*N_, (long)DH_*N_,
      nullptr, ctx, D_, (long)N_*D_, (long)DH_, nullptr,
      nullptr, 0, 0, 0, N_, DH_, N_, H_, 1.0f);

  // 10) out_x_pre = ctx @ wo + bo + x  (f32)
  gemm_bf16_k<0,0,true,4><<<dim3(D_/64, MX_/128, 1), blk, 0, stream>>>(
      ctx, D_, 0, 0, woT, D_, 0, 0, oxp, nullptr, D_, 0, 0, bo,
      x, D_, 0, 0, MX_, D_, D_, 1, 1.0f);

  // 11) node FFN: LN -> GEMM+GELU -> GEMM + residual -> d_out node region
  ln256_bf16_k<<<MX_/8, blk, 0, stream>>>(oxp, ffx_ln_g, ffx_ln_b, hx, MX_);
  gemm_bf16_k<1,1,false,4><<<dim3(DFF_/64, MX_/128, 1), blk, 0, stream>>>(
      hx, D_, 0, 0, fw1T, D_, 0, 0, nullptr, h1, DFF_, 0, 0, ffx_b1,
      nullptr, 0, 0, 0, MX_, DFF_, D_, 1, 1.0f);
  gemm_bf16_k<0,0,true,4><<<dim3(D_/64, MX_/128, 1), blk, 0, stream>>>(
      h1, DFF_, 0, 0, fw2T, DFF_, 0, 0, out_x, nullptr, D_, 0, 0, ffx_b2,
      oxp, D_, 0, 0, MX_, D_, DFF_, 1, 1.0f);

  // 12) fused edge FFN, in place on out_e
  edge_ffn_k<<<NE_/128, blk, 0, stream>>>(ffe_ln_g, ffe_ln_b, ew1T, ffe_b1,
                                          ew2T, ffe_b2, out_e);
}
